// HGNN_AC_40570261078605
// MI455X (gfx1250) — matrix operand order repaired and checked by reading, not compiled, not run on hardware
//
#include <hip/hip_runtime.h>
#include <hip/hip_bf16.h>

typedef __attribute__((ext_vector_type(16))) _Float16 v16h;
typedef __attribute__((ext_vector_type(8)))  float    v8f;

#define IN_DIM 512
#define HID    512
#define HEADS  8
#define NDST   4096
#define NSRC   4096

// ---------------------------------------------------------------------------
// WMMA fragment loaders (contiguous k order, same in A and B, wave32)
// A 16x32: lane (l,hi): a[2j+t] = A[l][ hi*16 + 2j + t ]
// B 32x16 from k-major tile BT[n][k]: b[2j+t] = BT[l][ hi*16 + 2j + t ]
// C/D: vgpr r -> (m = hi*8 + r, n = l)
// ---------------------------------------------------------------------------
__device__ __forceinline__ v16h load_a16(const _Float16* base, int lda, int l, int hi) {
    union { v16h v; unsigned u[8]; } r;
    const unsigned* p = (const unsigned*)base;
    const int row = l * lda;
#pragma unroll
    for (int j = 0; j < 8; ++j) {
        const int k = (hi << 4) + (j << 1);
        r.u[j] = p[(row + k) >> 1];
    }
    return r.v;
}

__device__ __forceinline__ v16h load_bT(const _Float16* bt, int ldb, int l, int hi) {
    union { v16h v; unsigned u[8]; } r;
    const unsigned* p = (const unsigned*)bt;
    const int row = l * ldb + (hi << 4);
#pragma unroll
    for (int j = 0; j < 8; ++j) r.u[j] = p[(row + (j << 1)) >> 1];
    return r.v;
}

__device__ __forceinline__ v8f vzero8() {
    v8f z;
#pragma unroll
    for (int i = 0; i < 8; ++i) z[i] = 0.f;
    return z;
}

__device__ __forceinline__ v8f wmma_f16(v16h a, v16h b, v8f c) {
    return __builtin_amdgcn_wmma_f32_16x16x32_f16(false, a, false, b, (short)0, c,
                                                  false, false);
}

__device__ __forceinline__ unsigned pack_h2(float x, float y) {
    union { _Float16 h[2]; unsigned u; } p;
    p.h[0] = (_Float16)x;
    p.h[1] = (_Float16)y;
    return p.u;
}

// Async global->LDS copy (CDNA5, ASYNCcnt). SADDR form: saddr base + 32-bit voffset.
__device__ __forceinline__ void async_b128(unsigned lds_off, const void* sbase,
                                           unsigned voff_bytes) {
    asm volatile("global_load_async_to_lds_b128 %0, %1, %2"
                 :: "v"(lds_off), "v"(voff_bytes),
                    "s"((unsigned long long)(size_t)sbase)
                 : "memory");
}

// ---------------------------------------------------------------------------
// Kernel 1: Q/K projections, f32 in -> f16 out (+bias), row-major [n][e]
// grid (NDST/128, HID/64, 2*HEADS), 256 threads (8 waves)
// ---------------------------------------------------------------------------
__global__ __launch_bounds__(256)
void proj_kernel(const float* __restrict__ emb_dest, const float* __restrict__ emb_src,
                 const float* __restrict__ Wq, const float* __restrict__ bq,
                 const float* __restrict__ Wk, const float* __restrict__ bk,
                 _Float16* __restrict__ qkv) {
    const int which = blockIdx.z >> 3;  // 0=q, 1=k
    const int h     = blockIdx.z & 7;
    const float* X = (which == 0) ? emb_dest : emb_src;
    const float* W = ((which == 0) ? Wq : Wk) + (size_t)h * IN_DIM * HID;
    const float* bias = ((which == 0) ? bq : bk) + h * HID;
    _Float16* out = qkv + ((size_t)which * HEADS + h) * (size_t)NDST * HID;

    const int m0 = blockIdx.x * 128;
    const int e0 = blockIdx.y * 64;

    __shared__ _Float16 As[128 * 32];  // X tile, k contiguous
    __shared__ _Float16 WT[64 * 32];   // W tile transposed: [e][k]

    const int tid = threadIdx.x;
    const int w = tid >> 5, lane = tid & 31;
    const int l = lane & 15, hi = lane >> 4;

    v8f acc[4];
#pragma unroll
    for (int n = 0; n < 4; ++n) acc[n] = vzero8();

    for (int k0 = 0; k0 < IN_DIM; k0 += 32) {
#pragma unroll
        for (int i = 0; i < 8; ++i) {  // 2048 k-pairs: float2 load, packed u32 store
            const int p = tid + i * 256;
            const int r = p >> 4, cp = p & 15;
            const float2 xv =
                ((const float2*)(X + (size_t)(m0 + r) * IN_DIM + k0))[cp];
            ((unsigned*)As)[r * 16 + cp] = pack_h2(xv.x, xv.y);
        }
#pragma unroll
        for (int i = 0; i < 4; ++i) {  // 1024 k-pairs of W^T (strided gather, packed)
            const int p = tid + i * 256;
            const int e = p & 63, kp = p >> 6;  // kp in [0,16)
            const float w0 = W[(size_t)(k0 + 2 * kp) * HID + e0 + e];
            const float w1 = W[(size_t)(k0 + 2 * kp + 1) * HID + e0 + e];
            ((unsigned*)WT)[e * 16 + kp] = pack_h2(w0, w1);
        }
        __syncthreads();

        const v16h a = load_a16(As + w * 16 * 32, 32, l, hi);
#pragma unroll
        for (int n = 0; n < 4; ++n) {
            const v16h b = load_bT(WT + n * 16 * 32, 32, l, hi);
            acc[n] = wmma_f16(a, b, acc[n]);
        }
        __syncthreads();
    }

#pragma unroll
    for (int n = 0; n < 4; ++n) {
        const int col = e0 + n * 16 + l;
        const float bb = bias[col];
#pragma unroll
        for (int r = 0; r < 8; ++r) {
            const int row = m0 + w * 16 + hi * 8 + r;
            out[(size_t)row * HID + col] = (_Float16)(acc[n][r] + bb);
        }
    }
}

// ---------------------------------------------------------------------------
// Kernel 1b: V projection with TRANSPOSED output: Vt[h][e][m] = (feat@Wv + bv)^T
// Swapped-operand GEMM: A[e][k] = Wv[k][e], B[k][m] = feat[m][k] (natural).
// grid (HID/128, NSRC/64, HEADS), 256 threads.
// ---------------------------------------------------------------------------
__global__ __launch_bounds__(256)
void projT_kernel(const float* __restrict__ feat_src,
                  const float* __restrict__ Wv, const float* __restrict__ bv,
                  _Float16* __restrict__ vt_out_base) {
    const int h = blockIdx.z;
    const float* W = Wv + (size_t)h * HID * HID;
    const float* bias = bv + h * HID;
    _Float16* out = vt_out_base + (size_t)h * HID * NSRC;  // [e][m]

    const int e0 = blockIdx.x * 128;  // M-dim = embedding col e
    const int n0 = blockIdx.y * 64;   // N-dim = src row m

    __shared__ _Float16 As[128 * 32]; // [e][k] = Wv transposed tile
    __shared__ _Float16 Bs[64 * 32];  // [m][k] = feat tile (natural layout)

    const int tid = threadIdx.x;
    const int w = tid >> 5, lane = tid & 31;
    const int l = lane & 15, hi = lane >> 4;

    v8f acc[4];
#pragma unroll
    for (int n = 0; n < 4; ++n) acc[n] = vzero8();

    for (int k0 = 0; k0 < HID; k0 += 32) {
#pragma unroll
        for (int i = 0; i < 8; ++i) {  // 2048 k-pairs of Wv^T (coalesced over e)
            const int p = tid + i * 256;
            const int er = p & 127, kp = p >> 7;  // kp in [0,16)
            const float w0 = W[(size_t)(k0 + 2 * kp) * HID + e0 + er];
            const float w1 = W[(size_t)(k0 + 2 * kp + 1) * HID + e0 + er];
            ((unsigned*)As)[er * 16 + kp] = pack_h2(w0, w1);
        }
#pragma unroll
        for (int i = 0; i < 4; ++i) {  // 1024 k-pairs of feat (k contiguous)
            const int p = tid + i * 256;
            const int mr = p >> 4, cp = p & 15;
            const float2 fv =
                ((const float2*)(feat_src + (size_t)(n0 + mr) * HID + k0))[cp];
            ((unsigned*)Bs)[mr * 16 + cp] = pack_h2(fv.x, fv.y);
        }
        __syncthreads();

        const v16h a = load_a16(As + w * 16 * 32, 32, l, hi);
#pragma unroll
        for (int n = 0; n < 4; ++n) {
            const v16h b = load_bT(Bs + n * 16 * 32, 32, l, hi);
            acc[n] = wmma_f16(a, b, acc[n]);
        }
        __syncthreads();
    }

#pragma unroll
    for (int n = 0; n < 4; ++n) {
#pragma unroll
        for (int r = 0; r < 8; ++r) {
            const int erow = e0 + w * 16 + hi * 8 + r;
            out[(size_t)erow * NSRC + n0 + n * 16 + l] =
                (_Float16)(acc[n][r] + bias[erow]);
        }
    }
}

// ---------------------------------------------------------------------------
// Kernel 2: flash attention. One (64-row dst strip, head) per block.
// 128 threads (4 waves); wave w owns dst rows [w*16, w*16+16).
// K tiles double-buffered + Vt tile via async LDS DMA overlapped with the
// S-phase WMMAs. Online softmax fully in registers (shfl_xor butterflies).
// O accumulator stored TRANSPOSED osT[col][row], stride 68 floats:
//   - 272 B col stride => every per-lane 8-row f32 run is float4-aligned
//   - stride 68 mod 64 banks = 4 => lanes hit distinct banks
// LDS 307456 B: qs 64K | ks0 32K | ks1 32K | vt 32K | osT 136K | ps 4K | lS
// ---------------------------------------------------------------------------
#define OS_STRIDE 68
#define QS_OFF  0u
#define KS0_OFF 65536u
#define KS1_OFF 98304u
#define VT_OFF  131072u
#define OS_OFF  163840u                       // 512*68*4 = 139264 B
#define PS_OFF  303104u
#define LS_OFF  307200u
#define SMEM_BYTES 307456

__global__ __launch_bounds__(128)
void attn_kernel(const _Float16* __restrict__ qkv, float* __restrict__ out) {
    extern __shared__ char smem[];
    _Float16* qs  = (_Float16*)(smem + QS_OFF);
    float*    osT = (float*)(smem + OS_OFF);
    _Float16* ps  = (_Float16*)(smem + PS_OFF);
    float*    lS  = (float*)(smem + LS_OFF);

    const _Float16* Qb  = qkv;
    const _Float16* Kb  = qkv + (size_t)HEADS * NDST * HID;
    const _Float16* Vtb = Kb + (size_t)HEADS * NSRC * HID;

    const int tid = threadIdx.x;
    const int w = tid >> 5, lane = tid & 31;
    const int l = lane & 15, hi = lane >> 4;
    const int m0 = blockIdx.x * 64;
    const int h  = blockIdx.y;
    const float scale = 0.044194173824159216f;  // 1/sqrt(512)

    const _Float16* Qh  = Qb + ((size_t)h * NDST + m0) * HID;
    const _Float16* Kh  = Kb + (size_t)h * NSRC * HID;
    const _Float16* VTh = Vtb + (size_t)h * HID * NSRC;   // [e][m]

    {   // Q strip 64x512 f16, contiguous (synchronous b128 copy)
        const uint4* src = (const uint4*)Qh;
        uint4* dst = (uint4*)qs;
#pragma unroll 4
        for (int i = 0; i < 32; ++i) dst[tid + i * 128] = src[tid + i * 128];
    }
    {   // zero O accumulator (512*68 f32 = 8704 float4)
        const float4 z = make_float4(0.f, 0.f, 0.f, 0.f);
        float4* od = (float4*)osT;
#pragma unroll 4
        for (int i = 0; i < 68; ++i) od[tid + i * 128] = z;
    }
    // prologue: async-load K tile 0 into buffer 0 (16 instr/wave)
#pragma unroll
    for (int i = 0; i < 16; ++i) {
        const unsigned idx = (unsigned)(tid + i * 128);
        async_b128(KS0_OFF + idx * 16u, Kh, idx * 16u);
    }

    // per-lane persistent softmax state for rows m = w*16 + hi*8 + r
    float mrow[8], lrow[8];
#pragma unroll
    for (int r = 0; r < 8; ++r) { mrow[r] = -1e30f; lrow[r] = 0.f; }

    for (int it = 0; it < NSRC / 32; ++it) {
        const int s0 = it * 32;
        const unsigned cur = (unsigned)(it & 1);
        const bool more = (it + 1 < NSRC / 32);

        // async-load Vt tile for THIS iteration (consumed after S phase):
        // vt[c*32 + s] = Vt[c][s0+s]; flat uint4 idx: c = idx>>2, j = idx&3
#pragma unroll
        for (int i = 0; i < 16; ++i) {
            const unsigned idx = (unsigned)(tid + i * 128);
            const unsigned c = idx >> 2, j = idx & 3;
            async_b128(VT_OFF + idx * 16u, VTh,
                       c * (NSRC * 2u) + (unsigned)s0 * 2u + j * 16u);
        }
        // async-load NEXT K tile into the other buffer
        if (more) {
#pragma unroll
            for (int i = 0; i < 16; ++i) {
                const unsigned idx = (unsigned)(tid + i * 128);
                async_b128((cur ? KS0_OFF : KS1_OFF) + idx * 16u, Kh,
                           (unsigned)(s0 + 32) * (HID * 2u) + idx * 16u);
            }
            // current K tile is older than the 32 newer ops; in-order completion
            asm volatile("s_wait_asynccnt 0x20" ::: "memory");
        } else {
            asm volatile("s_wait_asynccnt 0x10" ::: "memory");
        }
        __syncthreads();

        // ---- S = Q K^T : 16 rows x 32 src cols per wave ----
        const _Float16* ksb = (const _Float16*)(smem + (cur ? KS1_OFF : KS0_OFF));
        v8f acc0 = vzero8(), acc1 = vzero8();
        for (int k0 = 0; k0 < HID; k0 += 32) {
            const v16h a  = load_a16(qs + (size_t)(w * 16) * HID + k0, HID, l, hi);
            const v16h b0 = load_bT(ksb + k0, HID, l, hi);
            const v16h b1 = load_bT(ksb + (size_t)16 * HID + k0, HID, l, hi);
            acc0 = wmma_f16(a, b0, acc0);
            acc1 = wmma_f16(a, b1, acc1);
        }

        // ---- online softmax in registers (rows spread over 16-lane halves) ----
        _Float16* pw = ps + w * 512;
        float alpha[8];
#pragma unroll
        for (int r = 0; r < 8; ++r) {
            const float s0v = acc0[r] * scale, s1v = acc1[r] * scale;
            float mv = fmaxf(s0v, s1v);
#pragma unroll
            for (int off = 8; off; off >>= 1) mv = fmaxf(mv, __shfl_xor(mv, off, 32));
            const float newm = fmaxf(mrow[r], mv);
            alpha[r] = __expf(mrow[r] - newm);
            mrow[r] = newm;
            const float p0 = __expf(s0v - newm), p1 = __expf(s1v - newm);
            float su = p0 + p1;
#pragma unroll
            for (int off = 8; off; off >>= 1) su += __shfl_xor(su, off, 32);
            lrow[r] = lrow[r] * alpha[r] + su;
            pw[(hi * 8 + r) * 32 + l]      = (_Float16)p0;
            pw[(hi * 8 + r) * 32 + 16 + l] = (_Float16)p1;
        }

        // Vt tile must now be resident (only next-K ops may remain outstanding)
        if (more) asm volatile("s_wait_asynccnt 0x10" ::: "memory");
        else      asm volatile("s_wait_asynccnt 0x0"  ::: "memory");
        __syncthreads();

        // ---- O = alpha*O + P @ V (O transposed in LDS, float4 accesses) ----
        const _Float16* vt = (const _Float16*)(smem + VT_OFF);
        const v16h a = load_a16(pw, 32, l, hi);  // same-wave LDS RAW: DS in-order
        for (int nc = 0; nc < 32; ++nc) {
            const v16h b = load_bT(vt + (size_t)(nc * 16) * 32, 32, l, hi);
            float* ob = osT + (size_t)(nc * 16 + l) * OS_STRIDE + w * 16 + hi * 8;
            union { float4 f4[2]; float f[8]; } ov;
            ov.f4[0] = ((const float4*)ob)[0];
            ov.f4[1] = ((const float4*)ob)[1];
            v8f c;
#pragma unroll
            for (int r = 0; r < 8; ++r) c[r] = ov.f[r] * alpha[r];
            c = wmma_f16(a, b, c);
#pragma unroll
            for (int r = 0; r < 8; ++r) ov.f[r] = c[r];
            ((float4*)ob)[0] = ov.f4[0];
            ((float4*)ob)[1] = ov.f4[1];
        }
        __syncthreads();  // protect vt / ks buffers before next iteration's issues
    }

    if (l == 0) {
#pragma unroll
        for (int r = 0; r < 8; ++r) lS[w * 16 + hi * 8 + r] = lrow[r];
    }
    __syncthreads();

    // ---- epilogue: out += elu(O / l) / HEADS ----
    for (int i = 0; i < 256; ++i) {
        const int idx = tid + i * 128;  // == row*512 + col
        const int row = idx >> 9, col = idx & 511;
        float v = osT[(size_t)col * OS_STRIDE + row] / lS[row];
        v = (v > 0.f) ? v : (__expf(v) - 1.f);
        atomicAdd(out + (size_t)m0 * HID + idx, 0.125f * v);
    }
}

// ---------------------------------------------------------------------------
extern "C" void kernel_launch(void* const* d_in, const int* in_sizes, int n_in,
                              void* d_out, int out_size, void* d_ws, size_t ws_size,
                              hipStream_t stream) {
    const float* emb_dest = (const float*)d_in[0];
    const float* emb_src  = (const float*)d_in[1];
    const float* feat_src = (const float*)d_in[2];
    const float* Wq = (const float*)d_in[3];
    const float* bq = (const float*)d_in[4];
    const float* Wk = (const float*)d_in[5];
    const float* bk = (const float*)d_in[6];
    const float* Wv = (const float*)d_in[7];
    const float* bv = (const float*)d_in[8];
    float* out = (float*)d_out;
    _Float16* qkv = (_Float16*)d_ws;  // Q | K | Vt (f16): 3 * 32 MB = 96 MB

    hipMemsetAsync(d_out, 0, (size_t)NDST * HID * sizeof(float), stream);

    dim3 pgrid(NDST / 128, HID / 64, 2 * HEADS);
    proj_kernel<<<pgrid, 256, 0, stream>>>(emb_dest, emb_src, Wq, bq, Wk, bk, qkv);

    _Float16* vt_base = qkv + (size_t)2 * HEADS * NDST * HID;
    dim3 tgrid(HID / 128, NSRC / 64, HEADS);
    projT_kernel<<<tgrid, 256, 0, stream>>>(feat_src, Wv, bv, vt_base);

    hipFuncSetAttribute(reinterpret_cast<const void*>(attn_kernel),
                        hipFuncAttributeMaxDynamicSharedMemorySize, SMEM_BYTES);
    attn_kernel<<<dim3(NDST / 64, HEADS), 128, SMEM_BYTES, stream>>>(qkv, out);
}
